// SageNet_78924319031658
// MI455X (gfx1250) — matrix operand order repaired and checked by reading, not compiled, not run on hardware
//
#include <hip/hip_runtime.h>

#define N_NODES 100000
#define N_EDGES 800000
#define M_TILES (N_NODES / 16)   // 6250, exact

typedef __attribute__((ext_vector_type(16))) __bf16 v16bf;
typedef __attribute__((ext_vector_type(8)))  float  v8f;

static __device__ __forceinline__ unsigned short f2bf_bits(float f) {
    unsigned u = __builtin_bit_cast(unsigned, f);
    unsigned r = u + 0x7FFFu + ((u >> 16) & 1u);   // round-to-nearest-even
    return (unsigned short)(r >> 16);
}
static __device__ __forceinline__ __bf16 f2bf(float f) {
    return __builtin_bit_cast(__bf16, f2bf_bits(f));
}

// ---------------- graph preprocessing (CSR by dst) ---------------------------
__global__ void zero_i(int* p, int n) {
    int i = blockIdx.x * blockDim.x + threadIdx.x;
    if (i < n) p[i] = 0;
}
__global__ void count_deg(const int* __restrict__ dst, int* deg, int e) {
    int i = blockIdx.x * blockDim.x + threadIdx.x;
    if (i < e) atomicAdd(&deg[dst[i]], 1);
}
// single-block exclusive scan (1024 threads, serial over chunks)
__global__ void exscan(const int* __restrict__ in, int* __restrict__ out, int n) {
    __shared__ int smem[1024];
    __shared__ int carry;
    int tid = threadIdx.x;
    if (tid == 0) carry = 0;
    __syncthreads();
    for (int base = 0; base < n; base += 1024) {
        int v = (base + tid < n) ? in[base + tid] : 0;
        smem[tid] = v;
        __syncthreads();
        int val = v;
        for (int off = 1; off < 1024; off <<= 1) {
            int t = (tid >= off) ? smem[tid - off] : 0;
            __syncthreads();
            val += t;
            smem[tid] = val;
            __syncthreads();
        }
        int c = carry;                         // all threads read old carry
        if (base + tid < n) out[base + tid] = c + val - v;
        __syncthreads();
        if (tid == 1023) carry = c + smem[1023];
        __syncthreads();
    }
    if (tid == 0) out[n] = carry;
}
__global__ void make_inv_deg(const int* __restrict__ row_ptr, float* inv_deg, int n) {
    int i = blockIdx.x * blockDim.x + threadIdx.x;
    if (i < n) {
        int d = row_ptr[i + 1] - row_ptr[i];
        inv_deg[i] = d > 0 ? 1.0f / (float)d : 0.f;
    }
}
__global__ void fill_csr(const int* __restrict__ src, const int* __restrict__ dst,
                         const int* __restrict__ row_ptr, int* cursor,
                         int* __restrict__ csr_src, int e) {
    int i = blockIdx.x * blockDim.x + threadIdx.x;
    if (i < e) {
        int d = dst[i];
        int off = atomicAdd(&cursor[d], 1);
        csr_src[row_ptr[d] + off] = src[i];
    }
}

// ---------------- weight packer: [Wl|Wr] f32 -> per-lane bf16 B fragments ----
// frag f = kt*CT + ct ; element j -> B[kt*32 + (lane/16)*8 + (j&7) + (j>>3)*16][ct*16 + lane%16]
template<int DI, int DO, int DOP>
__global__ void pack_weights(const float* __restrict__ Wl, const float* __restrict__ Wr,
                             unsigned short* __restrict__ wp) {
    constexpr int CT = (2 * DOP) / 16;
    constexpr int KT = (DI + 31) / 32;
    int t = blockIdx.x * blockDim.x + threadIdx.x;    // one thread per (frag, lane)
    if (t >= KT * CT * 32) return;
    int lane = t & 31;
    int f    = t >> 5;
    int ct   = f % CT;
    int kt   = f / CT;
    int gc   = ct * 16 + (lane & 15);
    const float* W; int n;
    if (gc < DOP) { W = Wl; n = gc; } else { W = Wr; n = gc - DOP; }
    int kbase = kt * 32 + (lane >> 4) * 8;
    unsigned short* o = wp + (size_t)t * 16;
#pragma unroll
    for (int j = 0; j < 16; ++j) {
        int k = kbase + (j & 7) + (j >> 3) * 16;
        float v = (k < DI && n < DO) ? W[(size_t)k * DO + n] : 0.f;
        o[j] = f2bf_bits(v);
    }
}

// ---------------- WMMA GEMM with LDS-staged B --------------------------------
// C[N x CT*16] = A[N x DI] @ packedW ; B chunk (<=32KB) shared by 8 waves/block
template<int DI, int CT, bool RELU>
__global__ __launch_bounds__(256) void gemm_wmma(const float* __restrict__ A,
                                                 const unsigned short* __restrict__ wp,
                                                 float* __restrict__ C) {
    constexpr int KT   = (DI + 31) / 32;
    constexpr int KCH  = (32 / CT) < KT ? (32 / CT) : KT;   // kt per LDS chunk
    constexpr int LDC  = CT * 16;
    constexpr int SVEC = KCH * CT * 64;                     // uint4 per chunk
    __shared__ unsigned short smem[KCH * CT * 512];

    int wid    = blockIdx.x * 8 + (threadIdx.x >> 5);
    bool active = wid < M_TILES;
    int wave   = active ? wid : 0;                          // clamp, keep barriers valid
    int lane   = threadIdx.x & 31;
    int hi     = lane >> 4;
    int lo     = lane & 15;
    const float* arow = A + (size_t)(wave * 16 + lo) * DI;

    v8f acc[CT];
#pragma unroll
    for (int t = 0; t < CT; ++t) acc[t] = (v8f){};

    for (int ck = 0; ck < KT; ck += KCH) {
        __syncthreads();                                    // previous chunk fully consumed
        const uint4* g = (const uint4*)wp + (size_t)ck * CT * 64;
        uint4* s = (uint4*)smem;
        for (int v = threadIdx.x; v < SVEC; v += 256) s[v] = g[v];
        __syncthreads();

        for (int ktl = 0; ktl < KCH; ++ktl) {
            int kk = (ck + ktl) * 32;
            v16bf a;
            // run 0: K = kk + 8*hi + [0..7]  (always in-bounds)
            {
                const float4* p = (const float4*)(arow + kk + hi * 8);
                float4 f0 = p[0], f1 = p[1];
                float vals[8] = {f0.x, f0.y, f0.z, f0.w, f1.x, f1.y, f1.z, f1.w};
#pragma unroll
                for (int j = 0; j < 8; ++j) {
                    float v = vals[j];
                    if (RELU) v = v > 0.f ? v : 0.f;
                    a[j] = f2bf(v);
                }
            }
            // run 1: K = kk + 16 + 8*hi + [0..7]  (OOB only when DI==16 -> zeros)
            int b1 = kk + 16 + hi * 8;
            if (b1 + 8 <= DI) {
                const float4* p = (const float4*)(arow + b1);
                float4 f0 = p[0], f1 = p[1];
                float vals[8] = {f0.x, f0.y, f0.z, f0.w, f1.x, f1.y, f1.z, f1.w};
#pragma unroll
                for (int j = 0; j < 8; ++j) {
                    float v = vals[j];
                    if (RELU) v = v > 0.f ? v : 0.f;
                    a[8 + j] = f2bf(v);
                }
            } else {
#pragma unroll
                for (int j = 0; j < 8; ++j) a[8 + j] = f2bf(0.f);
            }
#pragma unroll
            for (int t = 0; t < CT; ++t) {
                v16bf b = *(const v16bf*)(smem + ((ktl * CT + t) * 32 + lane) * 16);
                acc[t] = __builtin_amdgcn_wmma_f32_16x16x32_bf16(
                    false, a, false, b, (short)0, acc[t], false, false);
            }
        }
    }
    if (active) {
#pragma unroll
        for (int t = 0; t < CT; ++t) {
#pragma unroll
            for (int v = 0; v < 8; ++v) {
                int row = wave * 16 + hi * 8 + v;
                C[(size_t)row * LDC + t * 16 + lo] = acc[t][v];
            }
        }
    }
}

// ------- fused gather+combine: agg = inv_deg*sum(hl[csr]) + hr + b -----------
template<int DO, int DOP>
__global__ void gather_combine(const int* __restrict__ row_ptr, const int* __restrict__ csr_src,
                               const float* __restrict__ hboth, const float* __restrict__ b,
                               const float* __restrict__ inv_deg, float* __restrict__ agg) {
    int i = blockIdx.x * blockDim.x + threadIdx.x;
    if (i >= N_NODES * DOP) return;
    int c = i & (DOP - 1);
    int n = i / DOP;
    int e0 = row_ptr[n], e1 = row_ptr[n + 1];
    float sum = 0.f;
    for (int j = e0; j < e1; ++j) {
        int s = csr_src[j];                                 // broadcast across DOP threads
        sum += hboth[(size_t)s * (2 * DOP) + c];            // coalesced row read
    }
    float bias = (c < DO) ? b[c] : 0.f;
    agg[i] = sum * inv_deg[n] + hboth[(size_t)n * (2 * DOP) + DOP + c] + bias;
}

// ---------------- final slice copy: out[N x 5] = agg[:, :5] ------------------
__global__ void final_copy(const float* __restrict__ agg, float* __restrict__ out) {
    int i = blockIdx.x * blockDim.x + threadIdx.x;
    if (i >= N_NODES * 5) return;
    int n = i / 5, c = i - n * 5;
    out[i] = agg[(size_t)n * 16 + c];
}

// ---------------- per-layer driver -------------------------------------------
template<int DI, int DO, int DOP, bool RELU_A>
static void run_layer(const float* A, const float* Wl, const float* Wr, const float* b,
                      const int* row_ptr, const int* csr_src, const float* inv_deg,
                      unsigned short* wp, float* hboth, float* agg, hipStream_t stream) {
    constexpr int CT = (2 * DOP) / 16;
    constexpr int KT = (DI + 31) / 32;
    pack_weights<DI, DO, DOP><<<(KT * CT * 32 + 255) / 256, 256, 0, stream>>>(Wl, Wr, wp);
    gemm_wmma<DI, CT, RELU_A><<<(M_TILES + 7) / 8, 256, 0, stream>>>(A, wp, hboth);
    gather_combine<DO, DOP><<<(N_NODES * DOP + 255) / 256, 256, 0, stream>>>(
        row_ptr, csr_src, hboth, b, inv_deg, agg);
}

extern "C" void kernel_launch(void* const* d_in, const int* in_sizes, int n_in,
                              void* d_out, int out_size, void* d_ws, size_t ws_size,
                              hipStream_t stream) {
    const float* x   = (const float*)d_in[0];
    const int*   ei  = (const int*)d_in[1];
    const int*   src = ei;
    const int*   dst = ei + N_EDGES;
    const float* W1l = (const float*)d_in[3],  *W1r = (const float*)d_in[4],  *b1 = (const float*)d_in[5];
    const float* W2l = (const float*)d_in[6],  *W2r = (const float*)d_in[7],  *b2 = (const float*)d_in[8];
    const float* W3l = (const float*)d_in[9],  *W3r = (const float*)d_in[10], *b3 = (const float*)d_in[11];
    const float* W4l = (const float*)d_in[12], *W4r = (const float*)d_in[13], *b4 = (const float*)d_in[14];

    // workspace layout (all offsets 256B aligned), total ~81.4 MB:
    char* ws = (char*)d_ws;
    float* inv_deg = (float*)(ws);                        //   400,000 B
    int*   row_ptr = (int*)(ws + 400128);                 //   400,004 B (N+1)
    int*   cursor  = (int*)(ws + 800256);                 //   400,000 B (deg, then cursor)
    int*   csr_src = (int*)(ws + 1200384);                // 3,200,000 B
    unsigned short* wp = (unsigned short*)(ws + 4400384); //   196,608 B
    float* hboth   = (float*)(ws + 4596992);              // 51,200,000 B (N x 128 max)
    float* agg     = (float*)(ws + 55796992);             // 25,600,000 B (N x 64 max)

    // ---- build CSR by destination (once; reused by all 4 layers) ----
    zero_i<<<(N_NODES + 255) / 256, 256, 0, stream>>>(cursor, N_NODES);
    count_deg<<<(N_EDGES + 255) / 256, 256, 0, stream>>>(dst, cursor, N_EDGES);
    exscan<<<1, 1024, 0, stream>>>(cursor, row_ptr, N_NODES);
    make_inv_deg<<<(N_NODES + 255) / 256, 256, 0, stream>>>(row_ptr, inv_deg, N_NODES);
    zero_i<<<(N_NODES + 255) / 256, 256, 0, stream>>>(cursor, N_NODES);
    fill_csr<<<(N_EDGES + 255) / 256, 256, 0, stream>>>(src, dst, row_ptr, cursor, csr_src, N_EDGES);

    // ---- 4 SAGE layers; agg (stride DOP == next DI) doubles as next input;
    //      ReLU deferred into next GEMM's A-fragment f32->bf16 conversion ----
    run_layer<768, 64, 64, false>(x,   W1l, W1r, b1, row_ptr, csr_src, inv_deg, wp, hboth, agg, stream);
    run_layer< 64, 32, 32, true >(agg, W2l, W2r, b2, row_ptr, csr_src, inv_deg, wp, hboth, agg, stream);
    run_layer< 32, 16, 16, true >(agg, W3l, W3r, b3, row_ptr, csr_src, inv_deg, wp, hboth, agg, stream);
    run_layer< 16,  5, 16, true >(agg, W4l, W4r, b4, row_ptr, csr_src, inv_deg, wp, hboth, agg, stream);

    final_copy<<<(N_NODES * 5 + 255) / 256, 256, 0, stream>>>(agg, (float*)d_out);
}
